// FullAttention_48361331752989
// MI455X (gfx1250) — matrix operand order repaired and checked by reading, not compiled, not run on hardware
//
#include <hip/hip_runtime.h>
#include <hip/hip_bf16.h>

#define DIM   768
#define NH    12
#define HD    64
#define BB    2
#define SEQ   4096
#define NQKV  (3 * DIM)   // 2304
#define MTOT  (BB * SEQ)  // 8192

typedef __attribute__((ext_vector_type(16))) __bf16 bf16x16;
typedef __attribute__((ext_vector_type(8)))  float  floatx8;
typedef unsigned short u16;

union Frag {
    bf16x16 v;
    uint4   q[2];
};

__device__ __forceinline__ u16 f2bf(float f) {
    __bf16 h = (__bf16)f;
    return __builtin_bit_cast(u16, h);
}

__device__ __forceinline__ floatx8 wmma_bf16(const Frag& a, const Frag& b, floatx8 c) {
    return __builtin_amdgcn_wmma_f32_16x16x32_bf16(false, a.v, false, b.v,
                                                   (short)0, c, false, false);
}

// ---------------- conversion kernels ----------------

__global__ void cvt_f32_bf16(const float* __restrict__ in, u16* __restrict__ out, int n) {
    int i = blockIdx.x * blockDim.x + threadIdx.x;
    if (i < n) out[i] = f2bf(in[i]);
}

// LDS-tiled transpose+convert: out[c][r] = bf16(in[r][c]).
// grid: (cols_in/32, rows_in/32), block 256. Both dims must be multiples of 32.
__global__ __launch_bounds__(256)
void cvt_transpose(const float* __restrict__ in, u16* __restrict__ out,
                   int rows_in, int cols_in) {
    __shared__ u16 tile[32][33];
    const int tx = threadIdx.x & 31;
    const int ty = threadIdx.x >> 5;          // 8 row-groups
    const int c0 = blockIdx.x * 32;           // input col base
    const int r0 = blockIdx.y * 32;           // input row base
#pragma unroll
    for (int i = ty; i < 32; i += 8)
        tile[i][tx] = f2bf(in[(size_t)(r0 + i) * cols_in + c0 + tx]);
    __syncthreads();
#pragma unroll
    for (int i = ty; i < 32; i += 8)
        out[(size_t)(c0 + i) * rows_in + r0 + tx] = tile[tx][i];
}

// ---------------- QKV projection ----------------
// Each wave computes a 32x64 tile (2 M-tiles x 4 N-tiles, 8 accumulators):
// per 32-K step: 2 A-frag + 4 B-frag loads feed 8 WMMAs (A reused x4, B x2).
// grid: (MTOT/128, NQKV/64), block 128 (4 waves stacked along M)
__global__ __launch_bounds__(128)
void qkv_gemm(const u16* __restrict__ xbf, const u16* __restrict__ wT,
              const float* __restrict__ bias,
              u16* __restrict__ Qb, u16* __restrict__ Kb, u16* __restrict__ Vt) {
    const int lane = threadIdx.x & 31;
    const int wave = threadIdx.x >> 5;
    const int half = lane >> 4;
    const int lrow = lane & 15;
    const int m0 = blockIdx.x * 128 + wave * 32;
    const int n0 = blockIdx.y * 64;

    floatx8 acc[2][4];
#pragma unroll
    for (int mt = 0; mt < 2; ++mt)
#pragma unroll
        for (int nt = 0; nt < 4; ++nt)
            acc[mt][nt] = (floatx8){0.f,0.f,0.f,0.f,0.f,0.f,0.f,0.f};

    const u16* ap0 = xbf + (size_t)(m0 + lrow) * DIM + half * 8;
    const u16* ap1 = ap0 + (size_t)16 * DIM;
    const u16* bp0 = wT + (size_t)(n0 + lrow) * DIM + half * 8;

#pragma unroll 2
    for (int k0 = 0; k0 < DIM; k0 += 32) {
        Frag a0, a1;
        a0.q[0] = *(const uint4*)(ap0 + k0);
        a0.q[1] = *(const uint4*)(ap0 + k0 + 16);
        a1.q[0] = *(const uint4*)(ap1 + k0);
        a1.q[1] = *(const uint4*)(ap1 + k0 + 16);
#pragma unroll
        for (int nt = 0; nt < 4; ++nt) {
            Frag b;
            const u16* bp = bp0 + (size_t)nt * 16 * DIM + k0;
            b.q[0] = *(const uint4*)(bp);
            b.q[1] = *(const uint4*)(bp + 16);
            acc[0][nt] = wmma_bf16(a0, b, acc[0][nt]);
            acc[1][nt] = wmma_bf16(a1, b, acc[1][nt]);
        }
    }

#pragma unroll
    for (int nt = 0; nt < 4; ++nt) {
        const int n = n0 + nt * 16 + lrow;
        const float bv = bias[n];
        const int which = n / DIM;      // 0=Q 1=K 2=V
        const int rem = n - which * DIM;
        const int h = rem >> 6;
        const int d = rem & 63;
#pragma unroll
        for (int mt = 0; mt < 2; ++mt) {
#pragma unroll
            for (int r = 0; r < 8; ++r) {
                int m = m0 + mt * 16 + r + half * 8;
                int bidx = m >> 12;             // / SEQ
                int s = m & (SEQ - 1);
                u16 val = f2bf(acc[mt][nt][r] + bv);
                size_t bh = (size_t)bidx * NH + h;
                if (which == 0)      Qb[(bh * SEQ + s) * HD + d] = val;
                else if (which == 1) Kb[(bh * SEQ + s) * HD + d] = val;
                else                 Vt[(bh * HD + d) * SEQ + s] = val;
            }
        }
    }
}

// ---------------- flash attention ----------------
// grid: (SEQ/64, BB*NH), block 128 (4 waves, each owns 16 query rows)
__global__ __launch_bounds__(128)
void flash_attn(const u16* __restrict__ Qb, const u16* __restrict__ Kb,
                const u16* __restrict__ Vt, u16* __restrict__ Abf) {
    __shared__ u16 Plds[4][16 * 32];

    const int lane = threadIdx.x & 31;
    const int wave = threadIdx.x >> 5;
    const int half = lane >> 4;
    const int lrow = lane & 15;
    const int bh = blockIdx.y;
    const int b = bh / NH;
    const int h = bh - b * NH;
    const int q0 = blockIdx.x * 64 + wave * 16;

    const u16* Qp = Qb + (size_t)bh * SEQ * HD;
    const u16* Kp = Kb + (size_t)bh * SEQ * HD;
    const u16* Vp = Vt + (size_t)bh * HD * SEQ;

    // Q fragments (A layout), two K-chunks over hd=64
    Frag qf0, qf1;
    {
        const u16* p = Qp + (size_t)(q0 + lrow) * HD;
        qf0.q[0] = *(const uint4*)(p + half * 16);
        qf0.q[1] = *(const uint4*)(p + 8 + half * 16);
        qf1.q[0] = *(const uint4*)(p + 32 + half * 16);
        qf1.q[1] = *(const uint4*)(p + 40 + half * 16);
    }

    floatx8 o0 = {0.f,0.f,0.f,0.f,0.f,0.f,0.f,0.f};
    floatx8 o1 = o0, o2 = o0, o3 = o0;
    float mrow[8], lsum[8];
#pragma unroll
    for (int r = 0; r < 8; ++r) { mrow[r] = -1e30f; lsum[r] = 0.f; }

    const int jmax = q0 + 15;
    u16* pl = Plds[wave];

    for (int j0 = 0; j0 <= jmax; j0 += 32) {
        // S = Q @ K^T for 32 keys (two 16-wide tiles)
        floatx8 s0 = {0.f,0.f,0.f,0.f,0.f,0.f,0.f,0.f};
        floatx8 s1 = s0;
        {
            const u16* kp0 = Kp + (size_t)(j0 + lrow) * HD + half * 16;
            const u16* kp1 = Kp + (size_t)(j0 + 16 + lrow) * HD + half * 16;
            Frag ka, kb;
            ka.q[0] = *(const uint4*)(kp0);      ka.q[1] = *(const uint4*)(kp0 + 8);
            kb.q[0] = *(const uint4*)(kp1);      kb.q[1] = *(const uint4*)(kp1 + 8);
            s0 = wmma_bf16(qf0, ka, s0);
            s1 = wmma_bf16(qf0, kb, s1);
            ka.q[0] = *(const uint4*)(kp0 + 32); ka.q[1] = *(const uint4*)(kp0 + 40);
            kb.q[0] = *(const uint4*)(kp1 + 32); kb.q[1] = *(const uint4*)(kp1 + 40);
            s0 = wmma_bf16(qf1, ka, s0);
            s1 = wmma_bf16(qf1, kb, s1);
        }

        // online softmax (rows live across the 16-lane half-groups)
        float p0[8], p1[8], alpha[8];
#pragma unroll
        for (int r = 0; r < 8; ++r) {
            int qrow = q0 + r + half * 8;
            float v0 = s0[r] * 0.125f;   // hd^-0.5
            float v1 = s1[r] * 0.125f;
            if (j0 + lrow > qrow)      v0 = -1e30f;
            if (j0 + 16 + lrow > qrow) v1 = -1e30f;
            float mx = fmaxf(v0, v1);
            mx = fmaxf(mx, __shfl_xor(mx, 1));
            mx = fmaxf(mx, __shfl_xor(mx, 2));
            mx = fmaxf(mx, __shfl_xor(mx, 4));
            mx = fmaxf(mx, __shfl_xor(mx, 8));
            float mi = fmaxf(mrow[r], mx);
            float e0 = __expf(v0 - mi);
            float e1 = __expf(v1 - mi);
            float ls = e0 + e1;
            ls += __shfl_xor(ls, 1);
            ls += __shfl_xor(ls, 2);
            ls += __shfl_xor(ls, 4);
            ls += __shfl_xor(ls, 8);
            float a = __expf(mrow[r] - mi);
            lsum[r] = lsum[r] * a + ls;
            mrow[r] = mi;
            alpha[r] = a;
            p0[r] = e0;
            p1[r] = e1;
        }
#pragma unroll
        for (int r = 0; r < 8; ++r) {
            o0[r] *= alpha[r]; o1[r] *= alpha[r];
            o2[r] *= alpha[r]; o3[r] *= alpha[r];
        }

        // D-layout P -> LDS -> A-layout fragment (wave-private slab; wave32
        // lockstep + in-order same-wave DS ops, so no block barrier — trip
        // counts differ per wave due to causality)
#pragma unroll
        for (int r = 0; r < 8; ++r) {
            int M = r + half * 8;
            pl[M * 32 + lrow]      = f2bf(p0[r]);
            pl[M * 32 + 16 + lrow] = f2bf(p1[r]);
        }
        __builtin_amdgcn_wave_barrier();
        asm volatile("" ::: "memory");
        Frag pf;
        pf.q[0] = *(const uint4*)(pl + lrow * 32 + half * 8);
        pf.q[1] = *(const uint4*)(pl + lrow * 32 + 16 + half * 8);
        asm volatile("" ::: "memory");
        __builtin_amdgcn_wave_barrier();

        // O += P @ V  (V transposed: contiguous along keys)
#pragma unroll
        for (int n = 0; n < 4; ++n) {
            const u16* vp = Vp + (size_t)(n * 16 + lrow) * SEQ + j0 + 8 * half;
            Frag vf;
            vf.q[0] = *(const uint4*)(vp);
            vf.q[1] = *(const uint4*)(vp + 16);
            floatx8 o = (n == 0) ? o0 : (n == 1) ? o1 : (n == 2) ? o2 : o3;
            o = wmma_bf16(pf, vf, o);
            if (n == 0) o0 = o; else if (n == 1) o1 = o; else if (n == 2) o2 = o; else o3 = o;
        }
    }

    // normalize and write attention output (recombine heads), bf16
    u16* outp = Abf + ((size_t)b * SEQ + q0) * DIM + h * HD;
#pragma unroll
    for (int r = 0; r < 8; ++r) {
        float rinv = 1.f / lsum[r];
        size_t row = (size_t)(r + half * 8) * DIM;
        outp[row + lrow]      = f2bf(o0[r] * rinv);
        outp[row + 16 + lrow] = f2bf(o1[r] * rinv);
        outp[row + 32 + lrow] = f2bf(o2[r] * rinv);
        outp[row + 48 + lrow] = f2bf(o3[r] * rinv);
    }
}

// ---------------- output projection ----------------
// Same 32x64-per-wave tiling as qkv_gemm.
// grid: (MTOT/128, DIM/64), block 128
__global__ __launch_bounds__(128)
void out_gemm(const u16* __restrict__ abf, const u16* __restrict__ wT,
              const float* __restrict__ bias, float* __restrict__ out) {
    const int lane = threadIdx.x & 31;
    const int wave = threadIdx.x >> 5;
    const int half = lane >> 4;
    const int lrow = lane & 15;
    const int m0 = blockIdx.x * 128 + wave * 32;
    const int n0 = blockIdx.y * 64;

    floatx8 acc[2][4];
#pragma unroll
    for (int mt = 0; mt < 2; ++mt)
#pragma unroll
        for (int nt = 0; nt < 4; ++nt)
            acc[mt][nt] = (floatx8){0.f,0.f,0.f,0.f,0.f,0.f,0.f,0.f};

    const u16* ap0 = abf + (size_t)(m0 + lrow) * DIM + half * 8;
    const u16* ap1 = ap0 + (size_t)16 * DIM;
    const u16* bp0 = wT + (size_t)(n0 + lrow) * DIM + half * 8;

#pragma unroll 2
    for (int k0 = 0; k0 < DIM; k0 += 32) {
        Frag a0, a1;
        a0.q[0] = *(const uint4*)(ap0 + k0);
        a0.q[1] = *(const uint4*)(ap0 + k0 + 16);
        a1.q[0] = *(const uint4*)(ap1 + k0);
        a1.q[1] = *(const uint4*)(ap1 + k0 + 16);
#pragma unroll
        for (int nt = 0; nt < 4; ++nt) {
            Frag b;
            const u16* bp = bp0 + (size_t)nt * 16 * DIM + k0;
            b.q[0] = *(const uint4*)(bp);
            b.q[1] = *(const uint4*)(bp + 16);
            acc[0][nt] = wmma_bf16(a0, b, acc[0][nt]);
            acc[1][nt] = wmma_bf16(a1, b, acc[1][nt]);
        }
    }

#pragma unroll
    for (int nt = 0; nt < 4; ++nt) {
        const int n = n0 + nt * 16 + lrow;
        const float bv = bias[n];
#pragma unroll
        for (int mt = 0; mt < 2; ++mt) {
#pragma unroll
            for (int r = 0; r < 8; ++r) {
                int m = m0 + mt * 16 + r + half * 8;
                out[(size_t)m * DIM + n] = acc[mt][nt][r] + bv;
            }
        }
    }
}

// ---------------- launch ----------------

extern "C" void kernel_launch(void* const* d_in, const int* in_sizes, int n_in,
                              void* d_out, int out_size, void* d_ws, size_t ws_size,
                              hipStream_t stream) {
    const float* x    = (const float*)d_in[0];
    const float* Wqkv = (const float*)d_in[1];
    const float* bqkv = (const float*)d_in[2];
    const float* Wout = (const float*)d_in[3];
    const float* bout = (const float*)d_in[4];
    float* out = (float*)d_out;

    char* ws = (char*)d_ws;
    size_t off = 0;
    auto alloc = [&](size_t bytes) -> char* {
        char* p = ws + off;
        off += (bytes + 255) & ~(size_t)255;
        return p;
    };
    u16* xbf   = (u16*)alloc((size_t)MTOT * DIM * 2);   // 12.6 MB
    u16* wqkvT = (u16*)alloc((size_t)NQKV * DIM * 2);   //  3.5 MB
    u16* woutT = (u16*)alloc((size_t)DIM * DIM * 2);    //  1.2 MB
    u16* Qb    = (u16*)alloc((size_t)BB * NH * SEQ * HD * 2); // 12.6 MB
    u16* Kb    = (u16*)alloc((size_t)BB * NH * SEQ * HD * 2); // 12.6 MB
    u16* Vt    = (u16*)alloc((size_t)BB * NH * HD * SEQ * 2); // 12.6 MB
    u16* Abf   = (u16*)alloc((size_t)MTOT * DIM * 2);   // 12.6 MB
    (void)ws_size; // ~68 MB total required

    {
        int n = MTOT * DIM;
        cvt_f32_bf16<<<(n + 255) / 256, 256, 0, stream>>>(x, xbf, n);
    }
    cvt_transpose<<<dim3(NQKV / 32, DIM / 32), 256, 0, stream>>>(Wqkv, wqkvT, DIM, NQKV);
    cvt_transpose<<<dim3(DIM / 32, DIM / 32), 256, 0, stream>>>(Wout, woutT, DIM, DIM);

    qkv_gemm<<<dim3(MTOT / 128, NQKV / 64), 128, 0, stream>>>(xbf, wqkvT, bqkv, Qb, Kb, Vt);

    flash_attn<<<dim3(SEQ / 64, BB * NH), 128, 0, stream>>>(Qb, Kb, Vt, Abf);

    out_gemm<<<dim3(MTOT / 128, DIM / 64), 128, 0, stream>>>(Abf, woutT, bout, out);
}
